// PointerNet_69715909148893
// MI455X (gfx1250) — matrix operand order repaired and checked by reading, not compiled, not run on hardware
//
#include <hip/hip_runtime.h>

// Shapes (fixed by the reference)
constexpr int Bq = 4, Tq = 512, Iq = 512, Hq = 8, Dq = 512, Vq = 32000;
constexpr long long BTV = (long long)Bq * Tq * Vq;          // 65,536,000

typedef __attribute__((ext_vector_type(16))) _Float16 v16h;
typedef __attribute__((ext_vector_type(8)))  float    v8f;

// ---------------------------------------------------------------- utilities
__global__ void fill_zero_kernel(float* __restrict__ p, long long n) {
    long long g = (long long)blockIdx.x * blockDim.x + threadIdx.x;
    if (g < n) p[g] = 0.0f;
}

// ------------------------------------------------- 1) attn = mean over heads
__global__ void attn_mean_kernel(const float* __restrict__ heads,
                                 float* __restrict__ attn) {
    int g = blockIdx.x * blockDim.x + threadIdx.x;      // over B*T*I
    if (g >= Bq * Tq * Iq) return;
    int b  = g / (Tq * Iq);
    int ti = g % (Tq * Iq);
    const float* p = heads + (long long)b * Hq * Tq * Iq + ti;
    float s = 0.0f;
#pragma unroll
    for (int h = 0; h < Hq; ++h) s += p[(long long)h * Tq * Iq];
    attn[g] = s * 0.125f;
}

// ---- 2) context tile via WMMA, fused reduction into p_gen logit accumulator
// context[b] = attn[b] (T x I) @ enc[b] (I x D); each wave owns a 16x16 tile.
// Instead of storing context we accumulate  sum_n tile[m][n]*W_pgen[n]  into
// pgen_acc[b*T + m] (atomic f32 add).
__global__ void __launch_bounds__(32)
ctx_wmma_pgen_kernel(const float* __restrict__ attn,
                     const float* __restrict__ enc,
                     const float* __restrict__ W,     // W_pgen, first D entries used
                     float* __restrict__ pgen_acc) {
    const int lane = threadIdx.x;          // 0..31, wave32
    const int tn = blockIdx.x;             // D/16 = 32 tiles
    const int tm = blockIdx.y;             // T/16 = 32 tiles
    const int b  = blockIdx.z;
    const int h  = lane >> 4;              // lane half
    const int l  = lane & 15;
    const int m0 = tm * 16, n0 = tn * 16;

    const float* arow  = attn + (long long)(b * Tq + m0 + l) * Iq;  // row M=l(+0)
    const float* bbase = enc  + (long long)b * Iq * Dq;
    const int cn = n0 + l;                 // this lane's output column N

    v8f acc = {};
    for (int kb = 0; kb < Iq; kb += 32) {
        // ---- A fragment (16-bit A 16x32 layout):
        // half h=0: K = kb+{0..7, 16..23}; half h=1: K = kb+{8..15, 24..31}
        const float4* pa0 = (const float4*)(arow + kb + 8 * h);
        const float4* pa1 = (const float4*)(arow + kb + 16 + 8 * h);
        float4 x0 = pa0[0], x1 = pa0[1];
        float4 y0 = pa1[0], y1 = pa1[1];
        v16h A;
        A[0]  = (_Float16)x0.x; A[1]  = (_Float16)x0.y;
        A[2]  = (_Float16)x0.z; A[3]  = (_Float16)x0.w;
        A[4]  = (_Float16)x1.x; A[5]  = (_Float16)x1.y;
        A[6]  = (_Float16)x1.z; A[7]  = (_Float16)x1.w;
        A[8]  = (_Float16)y0.x; A[9]  = (_Float16)y0.y;
        A[10] = (_Float16)y0.z; A[11] = (_Float16)y0.w;
        A[12] = (_Float16)y1.x; A[13] = (_Float16)y1.y;
        A[14] = (_Float16)y1.z; A[15] = (_Float16)y1.w;

        // ---- B fragment: rows striped across lanes; same K order as A: half h=0: K = kb+{0..7, 16..23}; half h=1: K = kb+{8..15, 24..31}
        v16h Bf;
#pragma unroll
        for (int v = 0; v < 8; ++v) {
            int k0 = kb + 2 * v + 8 * h + (v >= 4 ? 8 : 0);
            Bf[2 * v]     = (_Float16)bbase[(long long)k0 * Dq + cn];
            Bf[2 * v + 1] = (_Float16)bbase[(long long)(k0 + 1) * Dq + cn];
        }

        acc = __builtin_amdgcn_wmma_f32_16x16x32_f16(
            /*neg_a=*/false, A, /*neg_b=*/false, Bf,
            /*c_mod=*/(short)0, acc, /*reuse_a=*/false, /*reuse_b=*/false);
    }

    // Fused p_gen partial: tile row m gets  sum_n acc * W[n].
    // C/D layout: VGPR v -> M = v + 8h, N = l.
    float wn = W[cn];
#pragma unroll
    for (int v = 0; v < 8; ++v) {
        float part = acc[v] * wn;
#pragma unroll
        for (int off = 1; off < 16; off <<= 1)
            part += __shfl_xor(part, off, 32);      // reduce within each 16-lane half
        if (l == 0)
            atomicAdd(&pgen_acc[b * Tq + m0 + v + 8 * h], part);
    }
}

// ------------------------------- 3) p_gen = sigmoid(ctx.w1 + dec.w2 + tar.w3 + b)
__global__ void __launch_bounds__(256)
pgen_kernel(const float* __restrict__ dec, const float* __restrict__ tar,
            const float* __restrict__ W, const float* __restrict__ b0,
            const float* __restrict__ pgen_acc, float* __restrict__ pgen_out) {
    int gt   = blockIdx.x * blockDim.x + threadIdx.x;
    int wave = gt >> 5;                     // one wave per (b,t); wave = b*T + t
    int lane = threadIdx.x & 31;
    if (wave >= Bq * Tq) return;
    const float* drow = dec + (long long)wave * Dq;
    const float* trow = tar + (long long)wave * Dq;
    float s = 0.0f;
    for (int d = lane; d < Dq; d += 32)
        s += drow[d] * W[Dq + d] + trow[d] * W[2 * Dq + d];
#pragma unroll
    for (int off = 16; off; off >>= 1) s += __shfl_xor(s, off, 32);
    if (lane == 0) {
        float logit = pgen_acc[wave] + s + b0[0];
        pgen_out[wave] = 1.0f / (1.0f + expf(-logit));
    }
}

// ------------------------------- 4) scatter-add attn into vocab slots (sparse)
__global__ void scatter_kernel(const float* __restrict__ attn,
                               const int* __restrict__ tok,
                               float* __restrict__ ptr_out) {
    int g = blockIdx.x * blockDim.x + threadIdx.x;      // over B*T*I
    if (g >= Bq * Tq * Iq) return;
    int b   = g / (Tq * Iq);
    int rem = g % (Tq * Iq);
    int t   = rem / Iq;
    int i   = rem % Iq;
    int v   = tok[b * Iq + i];
    if (v >= 0 && v < Vq)
        atomicAdd(ptr_out + (long long)(b * Tq + t) * Vq + v, attn[g]);
}

// --------------- 5) row softmax over V + gated mix with generator distribution
__global__ void __launch_bounds__(256)
softmax_mix_kernel(const float* __restrict__ gen,
                   const float* __restrict__ pgen,
                   float* __restrict__ ptr_out,
                   float* __restrict__ final_out) {
    __shared__ float red[256];
    const int row = blockIdx.x;            // b*T + t
    const int tid = threadIdx.x;
    const long long base = (long long)row * Vq;
    float* prow = ptr_out + base;
    const float* grow = gen + base;
    float* frow = final_out + base;

    // max (scattered values are >= 0; zeros abound, so start at 0)
    float m = 0.0f;
    for (int v = tid; v < Vq; v += 256) m = fmaxf(m, prow[v]);
    red[tid] = m; __syncthreads();
    for (int s = 128; s; s >>= 1) {
        if (tid < s) red[tid] = fmaxf(red[tid], red[tid + s]);
        __syncthreads();
    }
    m = red[0]; __syncthreads();

    // sum of exp (row re-read hits L2/WGP$)
    float sum = 0.0f;
    for (int v = tid; v < Vq; v += 256) sum += expf(prow[v] - m);
    red[tid] = sum; __syncthreads();
    for (int s = 128; s; s >>= 1) {
        if (tid < s) red[tid] += red[tid + s];
        __syncthreads();
    }
    const float rden = 1.0f / red[0];
    const float pg = pgen[row];
    const float og = 1.0f - pg;

    for (int v = tid; v < Vq; v += 256) {
        float p = expf(prow[v] - m) * rden;
        prow[v] = p;                                   // pointer_output
        frow[v] = pg * grow[v] + og * p;               // final_output
    }
}

// --------------------------------------------------------------------- launch
extern "C" void kernel_launch(void* const* d_in, const int* in_sizes, int n_in,
                              void* d_out, int out_size, void* d_ws, size_t ws_size,
                              hipStream_t stream) {
    const int*   tok   = (const int*)  d_in[0];   // inp_tokens (B,I)
    const float* tar   = (const float*)d_in[1];   // tar_embedded (B,T,D)
    const float* gen   = (const float*)d_in[2];   // generator_output (B,T,V)
    const float* enc   = (const float*)d_in[3];   // enc_output (B,I,D)
    const float* dec   = (const float*)d_in[4];   // dec_state (B,T,D)
    const float* heads = (const float*)d_in[5];   // attn_heads (B,H,T,I)
    const float* W     = (const float*)d_in[6];   // W_pgen (3D,1)
    const float* b0    = (const float*)d_in[7];   // b_pgen (1,)

    float* out       = (float*)d_out;
    float* final_out = out;                       // (B,T,V)
    float* ptr_out   = out + BTV;                 // (B,T,V)
    float* pgen_out  = out + 2 * BTV;             // (B,T)

    float* attn     = (float*)d_ws;                        // B*T*I f32 (4 MB)
    float* pgen_acc = attn + (long long)Bq * Tq * Iq;      // B*T f32 (8 KB)

    // zero the scatter target and the logit accumulator
    fill_zero_kernel<<<(unsigned)((BTV + 255) / 256), 256, 0, stream>>>(ptr_out, BTV);
    fill_zero_kernel<<<(Bq * Tq + 255) / 256, 256, 0, stream>>>(pgen_acc, Bq * Tq);

    // 1) mean over heads
    attn_mean_kernel<<<(Bq * Tq * Iq) / 256, 256, 0, stream>>>(heads, attn);

    // 2) WMMA context GEMM fused with p_gen logit accumulation
    ctx_wmma_pgen_kernel<<<dim3(Dq / 16, Tq / 16, Bq), 32, 0, stream>>>(
        attn, enc, W, pgen_acc);

    // 3) p_gen gate
    pgen_kernel<<<(Bq * Tq * 32) / 256, 256, 0, stream>>>(
        dec, tar, W, b0, pgen_acc, pgen_out);

    // 4) sparse scatter of attention mass into vocab slots
    scatter_kernel<<<(Bq * Tq * Iq) / 256, 256, 0, stream>>>(attn, tok, ptr_out);

    // 5) row softmax + gated mixture (one workgroup per (b,t) row)
    softmax_mix_kernel<<<Bq * Tq, 256, 0, stream>>>(gen, pgen_out, ptr_out, final_out);
}
